// GraphTransformerBlock_30580167147521
// MI455X (gfx1250) — matrix operand order repaired and checked by reading, not compiled, not run on hardware
//
#include <hip/hip_runtime.h>
#include <hip/hip_bf16.h>
#include <math.h>

// ---------------------------------------------------------------------------
// Types for CDNA5 WMMA (wave32, v_wmma_f32_16x16x32_bf16)
// ---------------------------------------------------------------------------
typedef unsigned short u16;
typedef __attribute__((ext_vector_type(16))) __bf16 v16bf;
typedef __attribute__((ext_vector_type(8)))  float  v8f;
typedef __attribute__((ext_vector_type(8)))  u16    v8u;
typedef __attribute__((ext_vector_type(16))) u16    v16u;

#define D_MODEL 256
#define NHEAD   8
#define DK      32
#define DFF     1024

__device__ __forceinline__ u16 f32_to_bf16(float f) {
    unsigned x = __builtin_bit_cast(unsigned, f);
    unsigned r = x + 0x7FFFu + ((x >> 16) & 1u);          // round-to-nearest-even
    if ((x & 0x7F800000u) == 0x7F800000u) r = x;          // inf/nan passthrough
    return (u16)(r >> 16);
}

__device__ __forceinline__ v16bf asbf16(v16u x) {
    union { v16u u; v16bf b; } c; c.u = x; return c.b;
}

__device__ __forceinline__ float apply_act(float x, int act) {
    if (act == 1) {                                       // softplus (stable)
        return fmaxf(x, 0.f) + log1pf(expf(-fabsf(x)));
    } else if (act == 2) {                                // exact GELU
        return 0.5f * x * (1.f + erff(x * 0.70710678118654752440f));
    }
    return x;
}

// ---------------------------------------------------------------------------
// Conversion kernels
// ---------------------------------------------------------------------------
__global__ void cvt_f32_bf16(const float* __restrict__ in, u16* __restrict__ out, long n) {
    long i = (long)blockIdx.x * blockDim.x + threadIdx.x;
    if (i < n) out[i] = f32_to_bf16(in[i]);
}

// W f32 row-major [K, Nc]  ->  Wt bf16 column-major [Nc, K]
__global__ void cvt_transpose(const float* __restrict__ W, u16* __restrict__ Wt, int K, int Nc) {
    int i = blockIdx.x * blockDim.x + threadIdx.x;
    if (i < K * Nc) {
        int k = i / Nc, n = i % Nc;
        Wt[(size_t)n * K + k] = f32_to_bf16(W[i]);
    }
}

__global__ void zero_f32(float* __restrict__ p, long n) {
    long i = (long)blockIdx.x * blockDim.x + threadIdx.x;
    if (i < n) p[i] = 0.f;
}

// ---------------------------------------------------------------------------
// WMMA bf16 GEMM:  C[M,Nc] = act(A[M,K] @ B[K,Nc] + bias)
// A: bf16 row-major.  Bt: bf16 column-major ([Nc,K]).
// Block = 256 threads = 8 waves; block tile 128x64; wave tile 16x64.
// B (weights, <=0.5 MB) stays hot in WGP$/L2, so no B prefetch; the streaming
// A operand is prefetched one K-step ahead at WGP scope (locality 3).
// ---------------------------------------------------------------------------
__global__ void __launch_bounds__(256)
gemm_bf16(const u16* __restrict__ A, const u16* __restrict__ Bt,
          const float* __restrict__ bias, float* __restrict__ Cf,
          u16* __restrict__ Cb, int M, int K, int Nc, int act)
{
    const int wave = threadIdx.x >> 5;
    const int lane = threadIdx.x & 31;
    const int lo16 = lane & 15;
    const int hi   = lane >> 4;                 // 0 or 1 (lane half)
    const int m0 = blockIdx.x * 128 + wave * 16;
    const int n0 = blockIdx.y * 64;
    if (m0 >= M) return;                        // wave-uniform: EXEC stays all-ones

    int arow = m0 + lo16;
    if (arow > M - 1) arow = M - 1;             // clamp loads; stores masked below

    // A 16-bit layout: lanes 0-15 hold K = k0+0..15 (rows m0..m0+15),
    //                  lanes 16-31 hold K = k0+16..31.
    const u16* Aptr  = A  + (size_t)arow * K + hi * 16;
    // B 16-bit layout: lanes 0-15 hold K = k0+0..15 (col n), lanes 16-31 K = k0+16..31.
    const u16* Bptr0 = Bt + (size_t)(n0 +  0 + lo16) * K + hi * 16;
    const u16* Bptr1 = Bt + (size_t)(n0 + 16 + lo16) * K + hi * 16;
    const u16* Bptr2 = Bt + (size_t)(n0 + 32 + lo16) * K + hi * 16;
    const u16* Bptr3 = Bt + (size_t)(n0 + 48 + lo16) * K + hi * 16;

    v8f acc0 = {}, acc1 = {}, acc2 = {}, acc3 = {};

    for (int k0 = 0; k0 < K; k0 += 32) {
        // WGP-scope prefetch of next K-step of the streaming A operand
        __builtin_prefetch(Aptr + k0 + 32, 0, 3);

        v16u au;
        {
            v8u a0 = *(const v8u*)(Aptr + k0);
            v8u a1 = *(const v8u*)(Aptr + k0 + 8);
#pragma unroll
            for (int i = 0; i < 8; ++i) { au[i] = a0[i]; au[8 + i] = a1[i]; }
        }
        v16bf af = asbf16(au);

        v16bf bf0 = asbf16(*(const v16u*)(Bptr0 + k0));
        v16bf bf1 = asbf16(*(const v16u*)(Bptr1 + k0));
        v16bf bf2 = asbf16(*(const v16u*)(Bptr2 + k0));
        v16bf bf3 = asbf16(*(const v16u*)(Bptr3 + k0));

        acc0 = __builtin_amdgcn_wmma_f32_16x16x32_bf16(false, af, false, bf0, (short)0, acc0, false, false);
        acc1 = __builtin_amdgcn_wmma_f32_16x16x32_bf16(false, af, false, bf1, (short)0, acc1, false, false);
        acc2 = __builtin_amdgcn_wmma_f32_16x16x32_bf16(false, af, false, bf2, (short)0, acc2, false, false);
        acc3 = __builtin_amdgcn_wmma_f32_16x16x32_bf16(false, af, false, bf3, (short)0, acc3, false, false);
    }

    // C layout: VGPR r -> row m0 + r + 8*hi ; column = n-subtile base + (lane&15)
#pragma unroll
    for (int r = 0; r < 8; ++r) {
        int rr = m0 + r + hi * 8;
        if (rr < M) {
            size_t rowoff = (size_t)rr * Nc;
#pragma unroll
            for (int j = 0; j < 4; ++j) {
                float v = (j == 0 ? acc0[r] : j == 1 ? acc1[r] : j == 2 ? acc2[r] : acc3[r]);
                int col = n0 + 16 * j + lo16;
                v = apply_act(v + bias[col], act);
                if (Cf) Cf[rowoff + col] = v;
                if (Cb) Cb[rowoff + col] = f32_to_bf16(v);
            }
        }
    }
}

// ---------------------------------------------------------------------------
// Edge score kernel: one block per edge, one wave per head (DK==warpSize==32).
// s[e,h] = -sum_k Em[e,h,k] / sqrt(DK)
// ---------------------------------------------------------------------------
__global__ void __launch_bounds__(256)
edge_scores(const int* __restrict__ ei,
            const float* __restrict__ Qm, const float* __restrict__ Qv,
            const float* __restrict__ Km, const float* __restrict__ Kv,
            float* __restrict__ s, int E)
{
    int e = blockIdx.x;
    int h = threadIdx.x >> 5;
    int lane = threadIdx.x & 31;
    int src = ei[e];
    int dst = ei[E + e];
    size_t qoff = (size_t)dst * D_MODEL + h * DK + lane;
    size_t koff = (size_t)src * D_MODEL + h * DK + lane;
    float qm = Qm[qoff], qv = Qv[qoff];
    float km = Km[koff], kv = Kv[koff];
    float dmu = qm - km;
    float ratio = qv / kv;
    float em = dmu * dmu / (2.f * kv) + 0.5f * (ratio - logf(ratio) - 1.f);
#pragma unroll
    for (int m = 16; m > 0; m >>= 1) em += __shfl_xor(em, m, 32);
    if (lane == 0) s[(size_t)e * NHEAD + h] = -em * 0.17677669529663688f;  // 1/sqrt(32)
}

// ---------------------------------------------------------------------------
// Per-head softmax stats over ALL edges (faithful to reference: softmax axis=0)
// One block per head; max pass then sum-of-exp pass.
// ---------------------------------------------------------------------------
__global__ void head_stats(const float* __restrict__ s, float* __restrict__ hmax,
                           float* __restrict__ hsum, int E)
{
    int h = blockIdx.x;
    __shared__ float red[1024];
    float m = -3.4e38f;
    for (int e = threadIdx.x; e < E; e += blockDim.x)
        m = fmaxf(m, s[(size_t)e * NHEAD + h]);
    red[threadIdx.x] = m; __syncthreads();
    for (int w = 512; w > 0; w >>= 1) {
        if ((int)threadIdx.x < w) red[threadIdx.x] = fmaxf(red[threadIdx.x], red[threadIdx.x + w]);
        __syncthreads();
    }
    float mx = red[0]; __syncthreads();
    float sum = 0.f;
    for (int e = threadIdx.x; e < E; e += blockDim.x)
        sum += expf(s[(size_t)e * NHEAD + h] - mx);
    red[threadIdx.x] = sum; __syncthreads();
    for (int w = 512; w > 0; w >>= 1) {
        if ((int)threadIdx.x < w) red[threadIdx.x] += red[threadIdx.x + w];
        __syncthreads();
    }
    if (threadIdx.x == 0) { hmax[h] = mx; hsum[h] = red[0]; }
}

// ---------------------------------------------------------------------------
// Message scatter (segment_sum via f32 atomics). One block per edge; thread d
// handles feature d; head = d/32.
// ---------------------------------------------------------------------------
__global__ void __launch_bounds__(256)
scatter_msg(const int* __restrict__ ei, const float* __restrict__ s,
            const float* __restrict__ hmax, const float* __restrict__ hsum,
            const float* __restrict__ Vm, const float* __restrict__ Vv,
            float* __restrict__ muo, float* __restrict__ varo, int E)
{
    int e = blockIdx.x;
    int d = threadIdx.x;
    int h = d >> 5;
    int src = ei[e];
    int dst = ei[E + e];
    float a = expf(s[(size_t)e * NHEAD + h] - hmax[h]) / hsum[h];
    float vm = Vm[(size_t)src * D_MODEL + d];
    float vv = Vv[(size_t)src * D_MODEL + d];
    atomicAdd(muo  + (size_t)dst * D_MODEL + d, vm * a);
    atomicAdd(varo + (size_t)dst * D_MODEL + d, vv * a * a);
}

// ---------------------------------------------------------------------------
// out = LayerNorm(x + y) * g + b.  One block (256 threads) per row, D=256.
// Optionally also writes bf16 copy for the next WMMA GEMM stage.
// ---------------------------------------------------------------------------
__global__ void __launch_bounds__(256)
add_ln(const float* __restrict__ x, const float* __restrict__ y,
       const float* __restrict__ g, const float* __restrict__ b,
       float* __restrict__ outf, u16* __restrict__ outbf)
{
    int row = blockIdx.x;
    int d = threadIdx.x;
    size_t idx = (size_t)row * D_MODEL + d;
    float t = x[idx] + y[idx];
    __shared__ float red[D_MODEL];
    red[d] = t; __syncthreads();
    for (int w = 128; w > 0; w >>= 1) {
        if (d < w) red[d] += red[d + w];
        __syncthreads();
    }
    float mean = red[0] * (1.f / D_MODEL); __syncthreads();
    float c = t - mean;
    red[d] = c * c; __syncthreads();
    for (int w = 128; w > 0; w >>= 1) {
        if (d < w) red[d] += red[d + w];
        __syncthreads();
    }
    float var = red[0] * (1.f / D_MODEL);
    float o = c * rsqrtf(var + 1e-5f) * g[d] + b[d];
    if (outf)  outf[idx]  = o;
    if (outbf) outbf[idx] = f32_to_bf16(o);
}

// ---------------------------------------------------------------------------
// Host-side orchestration
// ---------------------------------------------------------------------------
extern "C" void kernel_launch(void* const* d_in, const int* in_sizes, int n_in,
                              void* d_out, int out_size, void* d_ws, size_t ws_size,
                              hipStream_t stream)
{
    (void)n_in; (void)out_size; (void)ws_size;

    const float* mu  = (const float*)d_in[0];
    const float* var = (const float*)d_in[1];
    const int*   ei  = (const int*)d_in[2];
    const float* Wqm = (const float*)d_in[3];  const float* bqm = (const float*)d_in[4];
    const float* Wqv = (const float*)d_in[5];  const float* bqv = (const float*)d_in[6];
    const float* Wkm = (const float*)d_in[7];  const float* bkm = (const float*)d_in[8];
    const float* Wkv = (const float*)d_in[9];  const float* bkv = (const float*)d_in[10];
    const float* Wvm = (const float*)d_in[11]; const float* bvm = (const float*)d_in[12];
    const float* Wvv = (const float*)d_in[13]; const float* bvv = (const float*)d_in[14];
    const float* Wom = (const float*)d_in[15]; const float* bom = (const float*)d_in[16];
    const float* Wov = (const float*)d_in[17]; const float* bov = (const float*)d_in[18];
    const float* W1  = (const float*)d_in[19]; const float* b1  = (const float*)d_in[20];
    const float* W2  = (const float*)d_in[21]; const float* b2  = (const float*)d_in[22];
    const float* g1  = (const float*)d_in[23]; const float* be1 = (const float*)d_in[24];
    const float* g2  = (const float*)d_in[25]; const float* be2 = (const float*)d_in[26];
    float* outp = (float*)d_out;

    const int N = in_sizes[0] / D_MODEL;
    const int E = in_sizes[2] / 2;
    const long ND = (long)N * D_MODEL;
    const long NF = (long)N * DFF;

    // ---- workspace layout (aliased; all stage dependencies are sequential) ----
    char* base = (char*)d_ws;
    size_t off = 0;
    auto take = [&](size_t bytes) { size_t o = off; off += (bytes + 255) & ~(size_t)255; return o; };

    const size_t o_WqmT = take((size_t)D_MODEL * D_MODEL * 2);
    const size_t o_WqvT = take((size_t)D_MODEL * D_MODEL * 2);
    const size_t o_WkmT = take((size_t)D_MODEL * D_MODEL * 2);
    const size_t o_WkvT = take((size_t)D_MODEL * D_MODEL * 2);
    const size_t o_WvmT = take((size_t)D_MODEL * D_MODEL * 2);
    const size_t o_WvvT = take((size_t)D_MODEL * D_MODEL * 2);
    const size_t o_WomT = take((size_t)D_MODEL * D_MODEL * 2);
    const size_t o_WovT = take((size_t)D_MODEL * D_MODEL * 2);
    const size_t o_W1T  = take((size_t)D_MODEL * DFF * 2);
    const size_t o_W2T  = take((size_t)DFF * D_MODEL * 2);
    const size_t o_mu_bf  = take((size_t)ND * 2);   // later reused as mu_out_bf
    const size_t o_var_bf = take((size_t)ND * 2);   // later reused as var_out_bf
    const size_t o_Qm = take((size_t)ND * 4);       // later: mu_out
    const size_t o_Qv = take((size_t)ND * 4);       // later: var_out
    const size_t o_Km = take((size_t)ND * 4);       // later: attn_mu
    const size_t o_Kv = take((size_t)ND * 4);       // later: attn_var
    const size_t o_Vm = take((size_t)ND * 4);       // later: mu1
    const size_t o_Vv = take((size_t)ND * 4);       // later: var1, then ff
    size_t sc_bytes = (size_t)E * NHEAD * 4;
    if (sc_bytes < (size_t)ND * 2) sc_bytes = (size_t)ND * 2;
    const size_t o_scores = take(sc_bytes);         // later reused as mu1_bf
    const size_t o_hmax = take(256);
    const size_t o_hsum = take(256);
    const size_t o_hid_bf = take((size_t)NF * 2);

    u16*   mu_bf   = (u16*)(base + o_mu_bf);
    u16*   var_bf  = (u16*)(base + o_var_bf);
    float* Qm = (float*)(base + o_Qm);  float* Qv = (float*)(base + o_Qv);
    float* Km = (float*)(base + o_Km);  float* Kv = (float*)(base + o_Kv);
    float* Vm = (float*)(base + o_Vm);  float* Vv = (float*)(base + o_Vv);
    float* scores = (float*)(base + o_scores);
    float* hmax = (float*)(base + o_hmax);
    float* hsum = (float*)(base + o_hsum);
    // aliases
    float* mu_out  = Qm;                u16* mu_out_bf  = mu_bf;
    float* var_out = Qv;                u16* var_out_bf = var_bf;
    float* attn_mu  = Km;
    float* attn_var = Kv;
    float* mu1 = Vm;                    u16* mu1_bf = (u16*)(base + o_scores);
    float* var1 = Vv;
    float* ff = Vv;                     // after var2 is done
    u16*   hid_bf = (u16*)(base + o_hid_bf);

    auto wt = [&](const float* W, size_t o, int K, int Nc) {
        int n = K * Nc;
        cvt_transpose<<<(n + 255) / 256, 256, 0, stream>>>(W, (u16*)(base + o), K, Nc);
    };
    auto gemm = [&](const u16* Ab, size_t oBt, const float* bias, float* Cf, u16* Cb,
                    int M, int K, int Nc, int act) {
        dim3 grid((M + 127) / 128, Nc / 64);
        gemm_bf16<<<grid, 256, 0, stream>>>(Ab, (const u16*)(base + oBt), bias, Cf, Cb, M, K, Nc, act);
    };

    // 1) pack weights: f32 [K,Nc] -> bf16 [Nc,K]
    wt(Wqm, o_WqmT, D_MODEL, D_MODEL);  wt(Wqv, o_WqvT, D_MODEL, D_MODEL);
    wt(Wkm, o_WkmT, D_MODEL, D_MODEL);  wt(Wkv, o_WkvT, D_MODEL, D_MODEL);
    wt(Wvm, o_WvmT, D_MODEL, D_MODEL);  wt(Wvv, o_WvvT, D_MODEL, D_MODEL);
    wt(Wom, o_WomT, D_MODEL, D_MODEL);  wt(Wov, o_WovT, D_MODEL, D_MODEL);
    wt(W1,  o_W1T,  D_MODEL, DFF);      wt(W2,  o_W2T,  DFF, D_MODEL);

    // 2) activations to bf16
    cvt_f32_bf16<<<(ND + 255) / 256, 256, 0, stream>>>(mu,  mu_bf,  ND);
    cvt_f32_bf16<<<(ND + 255) / 256, 256, 0, stream>>>(var, var_bf, ND);

    // 3) node projections (WMMA)
    gemm(mu_bf,  o_WqmT, bqm, Qm, nullptr, N, D_MODEL, D_MODEL, 0);
    gemm(var_bf, o_WqvT, bqv, Qv, nullptr, N, D_MODEL, D_MODEL, 1);
    gemm(mu_bf,  o_WkmT, bkm, Km, nullptr, N, D_MODEL, D_MODEL, 0);
    gemm(var_bf, o_WkvT, bkv, Kv, nullptr, N, D_MODEL, D_MODEL, 1);
    gemm(mu_bf,  o_WvmT, bvm, Vm, nullptr, N, D_MODEL, D_MODEL, 0);
    gemm(var_bf, o_WvvT, bvv, Vv, nullptr, N, D_MODEL, D_MODEL, 1);

    // 4) per-edge KL-style scores
    edge_scores<<<E, 256, 0, stream>>>(ei, Qm, Qv, Km, Kv, scores, E);

    // 5) global per-head softmax stats
    head_stats<<<NHEAD, 1024, 0, stream>>>(scores, hmax, hsum, E);

    // 6) segment-sum accumulators
    zero_f32<<<(ND + 255) / 256, 256, 0, stream>>>(mu_out,  ND);
    zero_f32<<<(ND + 255) / 256, 256, 0, stream>>>(var_out, ND);

    // 7) weighted message scatter
    scatter_msg<<<E, 256, 0, stream>>>(ei, scores, hmax, hsum, Vm, Vv, mu_out, var_out, E);

    // 8) attention output projections (WMMA)
    cvt_f32_bf16<<<(ND + 255) / 256, 256, 0, stream>>>(mu_out,  mu_out_bf,  ND);
    cvt_f32_bf16<<<(ND + 255) / 256, 256, 0, stream>>>(var_out, var_out_bf, ND);
    gemm(mu_out_bf,  o_WomT, bom, attn_mu,  nullptr, N, D_MODEL, D_MODEL, 0);
    gemm(var_out_bf, o_WovT, bov, attn_var, nullptr, N, D_MODEL, D_MODEL, 1);

    // 9) residual + LayerNorm
    add_ln<<<N, 256, 0, stream>>>(mu,  attn_mu,  g1, be1, mu1,  mu1_bf);
    add_ln<<<N, 256, 0, stream>>>(var, attn_var, g1, be1, var1, nullptr);
    // var2 = LN(var1 + var1, g2, be2) -> second half of output
    add_ln<<<N, 256, 0, stream>>>(var1, var1, g2, be2, outp + ND, nullptr);

    // 10) FFN (WMMA): hidden = gelu(mu1 @ W1 + b1) ; ff = hidden @ W2 + b2
    gemm(mu1_bf, o_W1T, b1, nullptr, hid_bf, N, D_MODEL, DFF, 2);
    gemm(hid_bf, o_W2T, b2, ff, nullptr, N, DFF, D_MODEL, 0);

    // 11) mu2 = LN(mu1 + ff, g2, be2) -> first half of output
    add_ln<<<N, 256, 0, stream>>>(mu1, ff, g2, be2, outp, nullptr);
}
